// PointerNet_52776558133581
// MI455X (gfx1250) — matrix operand order repaired and checked by reading, not compiled, not run on hardware
//
#include <hip/hip_runtime.h>
#include <hip/hip_bf16.h>
#include <math.h>

// ---------------------------------------------------------------------------
// Pointer-generator network forward loss on gfx1250 (MI455X).
//  * All GEMMs use v_wmma_f32_16x16x32_f16 (f16 operands, f32 accum).
//  * All B-operand weights are pre-packed into WMMA-fragment-major layout:
//    each (k-chunk 32, n-tile 16) fragment is a contiguous 32-lane x 32-byte
//    block -> ldB is one aligned v16h load (2x global_load_b128) per lane.
//  * Encoder: 1 persistent workgroup per direction; cell state c kept in
//    registers across all 512 steps, h recirculated through LDS (f16).
//  * Decoder: per step = [cell+attention kernel] -> [grid vocab sum-exp
//    WMMA kernel] -> [loss gather kernel]. Gold probability assembled
//    analytically (no 50100-wide distribution materialized).
// ---------------------------------------------------------------------------

typedef __attribute__((ext_vector_type(16))) _Float16 v16h;
typedef __attribute__((ext_vector_type(8)))  float    v8f;

#define B_     32
#define TENC_  512
#define TDEC_  40
#define E_     128
#define H_     256
#define V_     50000
#define EPS_   1e-12f

__device__ __forceinline__ float sigf(float x) { return 1.f / (1.f + expf(-x)); }

// ---- WMMA fragment helpers (16x16x32 f16 -> f32) --------------------------
// A tile: rows m0..m0+15, k cols k0..k0+31 of row-major [M][lda] f16 matrix.
__device__ __forceinline__ v16h ldA(const _Float16* A, int lda, int k0, int lane) {
    int row = lane & 15;
    int kb  = (lane & 16) ? 8 : 0;
    const _Float16* p = A + (size_t)row * lda + k0 + kb;
    v16h a;
#pragma unroll
    for (int i = 0; i < 8; ++i) { a[i] = p[i]; a[i + 8] = p[16 + i]; }
    return a;
}
// B tile from fragment-packed weights: fragment (kc, nt) of a [K][N] matrix
// occupies 32 lanes * 16 halves contiguously; lane's halves = its 16 K-rows.
__device__ __forceinline__ v16h ldB_pk(const _Float16* P, int kc, int nt, int NT, int lane) {
    const _Float16* p = P + (((size_t)kc * NT + nt) * 32 + lane) * 16;
    return *(const v16h*)p;                 // 32B aligned -> 2x b128
}
__device__ __forceinline__ v8f wmma_acc(v16h a, v16h b, v8f c) {
    return __builtin_amdgcn_wmma_f32_16x16x32_f16(false, a, false, b, (short)0, c,
                                                  false, false);
}
__device__ __forceinline__ v8f frag_bias(float bv) {
    v8f c;
#pragma unroll
    for (int r = 0; r < 8; ++r) c[r] = bv;
    return c;
}

// ---------------------------------------------------------------------------
// Pack an f32 weight matrix into fragment-major f16 layout.
// Logical B is [K][N]; transposed!=0 means src is [N][K] (PyTorch x@W.T).
__global__ void pack_kernel(_Float16* dst, const float* src, int K, int N, int transposed) {
    int i = blockIdx.x * 256 + threadIdx.x;
    if (i >= K * N) return;
    int half_i = i & 15;
    int lane   = (i >> 4) & 31;
    int tile   = i >> 9;
    int NT     = N >> 4;
    int nt     = tile % NT;
    int kc     = tile / NT;
    int k = kc * 32 + ((lane >> 4) & 1) * 8 + (half_i & 7) + (half_i >> 3) * 16;
    int n = nt * 16 + (lane & 15);
    float v = transposed ? src[(size_t)n * K + k] : src[(size_t)k * N + n];
    dst[i] = (_Float16)v;
}
__global__ void cvt_f16_kernel(_Float16* dst, const float* src, int n) {
    int i = blockIdx.x * 256 + threadIdx.x;
    if (i < n) dst[i] = (_Float16)src[i];
}
// Gather article embeddings -> f16 [B,T,E].
__global__ void embA_kernel(_Float16* dst, const float* embed, const int* ids, int npos) {
    int i = blockIdx.x * 256 + threadIdx.x;
    if (i < npos * E_) {
        int p = i >> 7, e = i & 127;
        dst[i] = (_Float16)embed[(size_t)ids[p] * E_ + e];
    }
}

// ---------------------------------------------------------------------------
// Encoder LSTM: blockIdx.x = direction (0 fwd, 1 bwd). 1024 threads = 32 waves.
// Wave w: mtile = w>>4 (16 batch rows), j = w&15 (gate column tile).
// Owns i/f/g/o tiles of the same columns -> c state stays in registers.
__global__ __launch_bounds__(1024) void enc_lstm_kernel(
    const _Float16* __restrict__ embA,   // [B,T,E] f16
    const _Float16* __restrict__ W2,     // packed [2][K=384][N=1024] f16
    const float* __restrict__ b_f, const float* __restrict__ b_b,
    float* __restrict__ enc_out,         // [B,T,512]
    float* __restrict__ hfin,            // [B,512] (dir-concat)
    float* __restrict__ cfin)            // [B,512]
{
    __shared__ _Float16 A[32 * 392];     // [B][x(128)|h(256)] f16, stride 392
    const int dir = blockIdx.x;
    const int tid = threadIdx.x, lane = tid & 31, wv = tid >> 5;
    const int mt = wv >> 4, j = wv & 15;
    const int coln = lane & 15, r0 = (lane & 16) ? 8 : 0;
    const int colg = j * 16 + coln;
    const _Float16* W   = W2 + (size_t)dir * 384 * 1024;
    const float*    bia = dir ? b_b : b_f;
    const float bi = bia[colg], bf = bia[256 + colg], bg = bia[512 + colg], bo = bia[768 + colg];

    v8f cfrag = {}, hfrag = {};
    for (int idx = tid; idx < 32 * 256; idx += 1024)
        A[(idx >> 8) * 392 + 128 + (idx & 255)] = (_Float16)0.f;
    __syncthreads();

    for (int s = 0; s < TENC_; ++s) {
        int t = dir ? (TENC_ - 1 - s) : s;
        for (int idx = tid; idx < 32 * 128; idx += 1024) {
            int b = idx >> 7, e = idx & 127;
            A[b * 392 + e] = embA[((size_t)b * TENC_ + t) * E_ + e];
        }
        __syncthreads();

        v8f gi = frag_bias(bi), gf = frag_bias(bf), gg = frag_bias(bg), go = frag_bias(bo);
        for (int k = 0; k < 384; k += 32) {
            v16h a = ldA(A + mt * 16 * 392, 392, k, lane);
            int kc = k >> 5;
            gi = wmma_acc(a, ldB_pk(W, kc,      j, 64, lane), gi);
            gf = wmma_acc(a, ldB_pk(W, kc, 16 + j, 64, lane), gf);
            gg = wmma_acc(a, ldB_pk(W, kc, 32 + j, 64, lane), gg);
            go = wmma_acc(a, ldB_pk(W, kc, 48 + j, 64, lane), go);
        }
        __syncthreads();
#pragma unroll
        for (int r = 0; r < 8; ++r) {
            float cn = sigf(gf[r]) * cfrag[r] + sigf(gi[r]) * tanhf(gg[r]);
            cfrag[r] = cn;
            float hn = sigf(go[r]) * tanhf(cn);
            hfrag[r] = hn;
            int row = mt * 16 + r0 + r;
            A[row * 392 + 128 + colg] = (_Float16)hn;
            enc_out[((size_t)row * TENC_ + t) * 512 + dir * 256 + colg] = hn;
        }
        __syncthreads();
    }
#pragma unroll
    for (int r = 0; r < 8; ++r) {
        int row = mt * 16 + r0 + r;
        hfin[row * 512 + dir * 256 + colg] = hfrag[r];
        cfin[row * 512 + dir * 256 + colg] = cfrag[r];
    }
}

// ---------------------------------------------------------------------------
// ReduceState (relu([hf|hb] @ red_W + b)) + decoder-state init.
__global__ __launch_bounds__(1024) void reduce_state_kernel(
    const float* hfin, const float* cfin,
    const _Float16* redWh, const float* red_bh,
    const _Float16* redWc, const float* red_bc,
    float* h, float* c,
    const float* h_context, const float* coverage,
    float* ctx, float* cov, float* loss_acc)
{
    __shared__ _Float16 Ah[32 * 520];
    const int tid = threadIdx.x, lane = tid & 31, wv = tid >> 5;
    for (int pass = 0; pass < 2; ++pass) {
        const float* src = pass ? cfin : hfin;
        for (int idx = tid; idx < 32 * 512; idx += 1024)
            Ah[(idx >> 9) * 520 + (idx & 511)] = (_Float16)src[idx];
        __syncthreads();
        {
            int mt = wv >> 4, nt = wv & 15;
            int col = nt * 16 + (lane & 15), r0 = (lane & 16) ? 8 : 0;
            const _Float16* Bm = pass ? redWc : redWh;
            const float* bb = pass ? red_bc : red_bh;
            v8f acc = frag_bias(bb[col]);
            for (int k = 0; k < 512; k += 32)
                acc = wmma_acc(ldA(Ah + mt * 16 * 520, 520, k, lane),
                               ldB_pk(Bm, k >> 5, nt, 16, lane), acc);
            float* dst = pass ? c : h;
#pragma unroll
            for (int r = 0; r < 8; ++r)
                dst[(size_t)(mt * 16 + r0 + r) * 256 + col] = fmaxf(acc[r], 0.f);
        }
        __syncthreads();
    }
    for (int idx = tid; idx < 32 * 512; idx += 1024) {
        ctx[idx] = h_context[idx];
        cov[idx] = coverage[idx];
    }
    if (tid < 32) loss_acc[tid] = 0.f;
}

// ---------------------------------------------------------------------------
// enc_feat = enc_out_f16 [16384,512] @ att_Wh [512,512] (packed, no bias).
__global__ __launch_bounds__(256) void feat_kernel(
    const _Float16* __restrict__ ench, const _Float16* __restrict__ Wh,
    float* __restrict__ feat)
{
    int lane = threadIdx.x & 31, wv = threadIdx.x >> 5;
    int mt = blockIdx.x;                  // 0..1023 (rows of B*T)
    int nt = blockIdx.y * 8 + wv;         // 0..31
    const _Float16* A = ench + (size_t)mt * 16 * 512;
    v8f acc = {};
    for (int k = 0; k < 512; k += 32)
        acc = wmma_acc(ldA(A, 512, k, lane), ldB_pk(Wh, k >> 5, nt, 32, lane), acc);
    int col = nt * 16 + (lane & 15), r0 = (lane & 16) ? 8 : 0;
#pragma unroll
    for (int r = 0; r < 8; ++r)
        feat[((size_t)mt * 16 + r0 + r) * 512 + col] = acc[r];
}

// ---------------------------------------------------------------------------
// Decoder step: cell + attention + coverage + p_gen + out1. One workgroup.
__global__ __launch_bounds__(1024) void dec_step_kernel(
    int t,
    const float* __restrict__ embed, const int* __restrict__ sids,
    const float* __restrict__ smask_all, const float* __restrict__ amask,
    const float* __restrict__ xc_b,  const _Float16* __restrict__ xcW,
    const float* __restrict__ dec_b, const _Float16* __restrict__ Wdec,
    const float* __restrict__ att_b, const _Float16* __restrict__ attWs,
    const float* __restrict__ att_wc, const float* __restrict__ att_v,
    const float* __restrict__ enc_feat, const float* __restrict__ enc_out,
    const float* __restrict__ pgenW, const float* __restrict__ pgenb,
    const float* __restrict__ out1_b, const _Float16* __restrict__ out1W,
    float* h, float* c, float* ctx, float* cov, float* attn,
    float* x_in, float* out1h, _Float16* out1h_h,
    float* pgen, float* sumexp, float* loss_acc)
{
    // Shared memory plan (halves):
    //   [0 .. 24831]   : A0(stride648)/A1(stride392)/Ash(stride520)/eS(512)/Aout(776); A0/Aout overlap decfea
    //   [16640..33023] : decfea f16 [32][512]
    __shared__ __align__(16) _Float16 S[33024];
    const int tid = threadIdx.x, lane = tid & 31, wv = tid >> 5;
    const int coln = lane & 15, r0 = (lane & 16) ? 8 : 0;

    // ---- Phase A: x_in = [ctx | emb(y_t)] @ xc_W + xc_b -------------------
    for (int idx = tid; idx < 32 * 640; idx += 1024) {
        int b = idx / 640, ci = idx % 640;
        float v;
        if (ci < 512) v = ctx[b * 512 + ci];
        else          v = embed[(size_t)sids[b * TDEC_ + t] * E_ + (ci - 512)];
        S[b * 648 + ci] = (_Float16)v;
    }
    __syncthreads();
    v8f xacc = {};
    if (wv < 16) {
        int mt = wv >> 3, nt = wv & 7;
        xacc = frag_bias(xc_b[nt * 16 + coln]);
        for (int k = 0; k < 640; k += 32)
            xacc = wmma_acc(ldA(S + mt * 16 * 648, 648, k, lane),
                            ldB_pk(xcW, k >> 5, nt, 8, lane), xacc);
    }
    __syncthreads();
    if (wv < 16) {
        int mt = wv >> 3, nt = wv & 7, col = nt * 16 + coln;
#pragma unroll
        for (int r = 0; r < 8; ++r) {
            int row = mt * 16 + r0 + r;
            S[row * 392 + col]     = (_Float16)xacc[r];
            x_in[row * 128 + col]  = xacc[r];
        }
    }
    for (int idx = tid; idx < 32 * 256; idx += 1024)
        S[(idx >> 8) * 392 + 128 + (idx & 255)] = (_Float16)h[idx];
    __threadfence();
    __syncthreads();

    // ---- Phase B: LSTM cell ----------------------------------------------
    {
        int mt = wv >> 4, j = wv & 15, colg = j * 16 + coln;
        v8f gi = frag_bias(dec_b[colg]),       gf = frag_bias(dec_b[256 + colg]);
        v8f gg = frag_bias(dec_b[512 + colg]), go = frag_bias(dec_b[768 + colg]);
        for (int k = 0; k < 384; k += 32) {
            v16h a = ldA(S + mt * 16 * 392, 392, k, lane);
            int kc = k >> 5;
            gi = wmma_acc(a, ldB_pk(Wdec, kc,      j, 64, lane), gi);
            gf = wmma_acc(a, ldB_pk(Wdec, kc, 16 + j, 64, lane), gf);
            gg = wmma_acc(a, ldB_pk(Wdec, kc, 32 + j, 64, lane), gg);
            go = wmma_acc(a, ldB_pk(Wdec, kc, 48 + j, 64, lane), go);
        }
        __syncthreads();
#pragma unroll
        for (int r = 0; r < 8; ++r) {
            int row = mt * 16 + r0 + r;
            float cold = c[row * 256 + colg];
            float cn = sigf(gf[r]) * cold + sigf(gi[r]) * tanhf(gg[r]);
            float hn = sigf(go[r]) * tanhf(cn);
            c[row * 256 + colg] = cn;
            h[row * 256 + colg] = hn;
            S[row * 520 + colg]       = (_Float16)hn;   // Ash: s_hat = [h|c]
            S[row * 520 + 256 + colg] = (_Float16)cn;
        }
    }
    __threadfence();
    __syncthreads();

    // ---- Phase C: dec_fea = s_hat @ att_Ws + att_b (f16 into LDS) --------
    _Float16* dfe = S + 16640;
    for (int p = 0; p < 2; ++p) {
        int tile = wv + 32 * p;            // 64 tiles
        int mt = tile >> 5, nt = tile & 31;
        int col = nt * 16 + coln;
        v8f acc = frag_bias(att_b[col]);
        for (int k = 0; k < 512; k += 32)
            acc = wmma_acc(ldA(S + mt * 16 * 520, 520, k, lane),
                           ldB_pk(attWs, k >> 5, nt, 32, lane), acc);
#pragma unroll
        for (int r = 0; r < 8; ++r)
            dfe[(mt * 16 + r0 + r) * 512 + col] = (_Float16)acc[r];
    }
    __syncthreads();

    // ---- Phase D: attention scores, softmax, coverage --------------------
    for (int idx = tid; idx < 32 * 512; idx += 1024) {
        int b = idx >> 9, tt = idx & 511;
        float cv = cov[b * 512 + tt];
        const float*    ef = enc_feat + ((size_t)(b * 512 + tt)) * 512;
        const _Float16* df = dfe + b * 512;
        float acc = 0.f;
        for (int d = 0; d < 512; ++d)
            acc += att_v[d] * tanhf(ef[d] + (float)df[d] + cv * att_wc[d]);
        S[b * 512 + tt] = (_Float16)acc;   // eS
    }
    __syncthreads();
    {
        int b = wv;
        float smk = smask_all[b * TDEC_ + t];
        float mx = -1e30f;
        for (int i = lane; i < 512; i += 32) mx = fmaxf(mx, (float)S[b * 512 + i]);
        for (int m = 16; m > 0; m >>= 1) mx = fmaxf(mx, __shfl_xor(mx, m, 32));
        float sm = 0.f;
        for (int i = lane; i < 512; i += 32) sm += expf((float)S[b * 512 + i] - mx);
        for (int m = 16; m > 0; m >>= 1) sm += __shfl_xor(sm, m, 32);
        float s2 = 0.f;
        for (int i = lane; i < 512; i += 32)
            s2 += (expf((float)S[b * 512 + i] - mx) / sm) * amask[b * 512 + i];
        for (int m = 16; m > 0; m >>= 1) s2 += __shfl_xor(s2, m, 32);
        float inv = 1.f / ((s2 + EPS_) * sm);
        float cl = 0.f;
        for (int i = lane; i < 512; i += 32) {
            float a = expf((float)S[b * 512 + i] - mx) * amask[b * 512 + i] * inv;
            float cvv = cov[b * 512 + i];
            cl += fminf(a, cvv);            // coverage loss uses cov BEFORE update
            attn[b * 512 + i] = a;
            cov[b * 512 + i]  = cvv + a;
        }
        for (int m = 16; m > 0; m >>= 1) cl += __shfl_xor(cl, m, 32);
        if (lane == 0) loss_acc[b] += cl * smk;   // COV_WT = 1.0
    }
    __threadfence();
    __syncthreads();

    // ---- Phase E: ctx_new = attn . enc_out -------------------------------
    for (int idx = tid; idx < 32 * 512; idx += 1024) {
        int b = idx >> 9, d = idx & 511;
        const float* eo = enc_out + (size_t)b * 512 * 512 + d;
        const float* at = attn + b * 512;
        float acc = 0.f;
        for (int tt = 0; tt < 512; ++tt) acc += at[tt] * eo[(size_t)tt * 512];
        ctx[b * 512 + d] = acc;
    }
    __threadfence();
    __syncthreads();

    // ---- Phase F: p_gen ---------------------------------------------------
    {
        int b = wv;
        float acc = 0.f;
        for (int d = lane; d < 1152; d += 32) {
            float v;
            if (d < 512)       v = ctx[b * 512 + d];
            else if (d < 768)  v = h[b * 256 + d - 512];
            else if (d < 1024) v = c[b * 256 + d - 768];
            else               v = x_in[b * 128 + d - 1024];
            acc += v * pgenW[d];
        }
        for (int m = 16; m > 0; m >>= 1) acc += __shfl_xor(acc, m, 32);
        if (lane == 0) pgen[b] = sigf(acc + pgenb[0]);
    }

    // ---- Phase G: out1 = [h | ctx] @ out1_W + out1_b ---------------------
    for (int idx = tid; idx < 32 * 768; idx += 1024) {
        int b = idx / 768, d = idx % 768;
        float v = (d < 256) ? h[b * 256 + d] : ctx[b * 512 + d - 256];
        S[b * 776 + d] = (_Float16)v;
    }
    __syncthreads();
    {
        int mt = wv >> 4, nt = wv & 15, col = nt * 16 + coln;
        v8f acc = frag_bias(out1_b[col]);
        for (int k = 0; k < 768; k += 32)
            acc = wmma_acc(ldA(S + mt * 16 * 776, 776, k, lane),
                           ldB_pk(out1W, k >> 5, nt, 16, lane), acc);
#pragma unroll
        for (int r = 0; r < 8; ++r) {
            int row = mt * 16 + r0 + r;
            out1h[row * 256 + col]   = acc[r];
            out1h_h[row * 256 + col] = (_Float16)acc[r];
        }
    }
    if (tid < 32) sumexp[tid] = 0.f;       // for the vocab kernel
}

// ---------------------------------------------------------------------------
// Vocab logits (32x50000) + per-row sum(exp) via atomics. 8 waves/block.
__global__ __launch_bounds__(256) void vocab_kernel(
    const _Float16* __restrict__ Ah,     // out1h f16 [32][256]
    const _Float16* __restrict__ W,      // out2_W packed [256][50000]
    const float* __restrict__ bias,      // out2_b [50000]
    float* __restrict__ sumexp)          // [32]
{
    int lane = threadIdx.x & 31, wv = threadIdx.x >> 5;
    int tile = blockIdx.x * 8 + wv;       // over 3125 col-tiles x 2 m-tiles
    int ct = tile >> 1, mt = tile & 1;
    if (ct >= V_ / 16) return;
    int col = ct * 16 + (lane & 15), r0 = (lane & 16) ? 8 : 0;
    v8f acc = frag_bias(bias[col]);
    for (int k = 0; k < 256; k += 32)
        acc = wmma_acc(ldA(Ah + mt * 16 * 256, 256, k, lane),
                       ldB_pk(W, k >> 5, ct, V_ / 16, lane), acc);
    float s[8];
#pragma unroll
    for (int r = 0; r < 8; ++r) s[r] = expf(acc[r]);
#pragma unroll
    for (int m = 1; m <= 8; m <<= 1)
#pragma unroll
        for (int r = 0; r < 8; ++r) s[r] += __shfl_xor(s[r], m, 32);
    if ((lane & 15) == 0)
#pragma unroll
        for (int r = 0; r < 8; ++r) atomicAdd(&sumexp[mt * 16 + r0 + r], s[r]);
}

// ---------------------------------------------------------------------------
// Per-step loss: gold = pgen*softmax[target] + (1-pgen)*sum(attn where copy id
// matches). Wave b handles batch row b.
__global__ __launch_bounds__(1024) void step_loss_kernel(
    int t, const int* __restrict__ tgt, const float* __restrict__ smask_all,
    const float* __restrict__ out1h, const float* __restrict__ out2W,
    const float* __restrict__ out2b, const float* __restrict__ pgen,
    const float* __restrict__ sumexp, const int* __restrict__ ext_ids,
    const float* __restrict__ attn, float* loss_acc)
{
    int lane = threadIdx.x & 31, b = threadIdx.x >> 5;
    int target = tgt[b * TDEC_ + t];
    float gold = 0.f;
    if (target < V_) {
        float acc = 0.f;
        for (int k = lane; k < 256; k += 32)
            acc += out1h[b * 256 + k] * out2W[(size_t)k * V_ + target];
        for (int m = 16; m > 0; m >>= 1) acc += __shfl_xor(acc, m, 32);
        gold = pgen[b] * expf(acc + out2b[target]) / sumexp[b];
    }
    float pa = 0.f;
    for (int j = lane; j < TENC_; j += 32)
        if (ext_ids[b * TENC_ + j] == target) pa += attn[b * TENC_ + j];
    for (int m = 16; m > 0; m >>= 1) pa += __shfl_xor(pa, m, 32);
    if (lane == 0)
        loss_acc[b] += -logf(gold + (1.f - pgen[b]) * pa + EPS_) * smask_all[b * TDEC_ + t];
}

__global__ void finalize_kernel(const float* loss_acc, const int* slen, float* out) {
    int lane = threadIdx.x;
    float v = (lane < 32) ? loss_acc[lane] / (float)slen[lane] : 0.f;
    for (int m = 16; m > 0; m >>= 1) v += __shfl_xor(v, m, 32);
    if (lane == 0) out[0] = v / 32.f;
}

// ---------------------------------------------------------------------------
extern "C" void kernel_launch(void* const* d_in, const int* in_sizes, int n_in,
                              void* d_out, int out_size, void* d_ws, size_t ws_size,
                              hipStream_t stream) {
    // Input order (setup_inputs dict, params flattened in insertion order):
    const int*   article_ids = (const int*)d_in[0];
    const float* amask       = (const float*)d_in[2];
    const int*   ext_ids     = (const int*)d_in[3];
    const int*   sids        = (const int*)d_in[4];
    const int*   tgt         = (const int*)d_in[5];
    const float* smask       = (const float*)d_in[6];
    const int*   slen        = (const int*)d_in[7];
    const float* h_context   = (const float*)d_in[8];
    const float* coverage    = (const float*)d_in[10];
    const float* embed       = (const float*)d_in[11];
    const float* enc_Wih_f = (const float*)d_in[12], *enc_Whh_f = (const float*)d_in[13];
    const float* enc_b_f   = (const float*)d_in[14];
    const float* enc_Wih_b = (const float*)d_in[15], *enc_Whh_b = (const float*)d_in[16];
    const float* enc_b_b   = (const float*)d_in[17];
    const float* red_Wh = (const float*)d_in[18], *red_bh = (const float*)d_in[19];
    const float* red_Wc = (const float*)d_in[20], *red_bc = (const float*)d_in[21];
    const float* xc_W = (const float*)d_in[22], *xc_b = (const float*)d_in[23];
    const float* dec_Wih = (const float*)d_in[24], *dec_Whh = (const float*)d_in[25];
    const float* dec_b = (const float*)d_in[26];
    const float* att_Wh = (const float*)d_in[27], *att_Ws = (const float*)d_in[28];
    const float* att_b = (const float*)d_in[29], *att_wc = (const float*)d_in[30];
    const float* att_v = (const float*)d_in[31];
    const float* pgen_W = (const float*)d_in[32], *pgen_b = (const float*)d_in[33];
    const float* out1_W = (const float*)d_in[34], *out1_b = (const float*)d_in[35];
    const float* out2_W = (const float*)d_in[36], *out2_b = (const float*)d_in[37];

    // Workspace bump allocator (deterministic layout; ~119 MB total).
    char* wsp = (char*)d_ws;
    size_t off = 0;
    auto alloc = [&](size_t bytes) -> void* {
        void* p = wsp + off;
        off += (bytes + 255) & ~(size_t)255;
        return p;
    };
    _Float16* embA    = (_Float16*)alloc((size_t)B_ * TENC_ * E_ * 2);
    _Float16* Wenc    = (_Float16*)alloc((size_t)2 * 384 * 1024 * 2);  // packed
    _Float16* Wdec    = (_Float16*)alloc((size_t)384 * 1024 * 2);      // packed
    _Float16* attWh_p = (_Float16*)alloc((size_t)512 * 512 * 2);
    _Float16* attWs_p = (_Float16*)alloc((size_t)512 * 512 * 2);
    _Float16* xcW_p   = (_Float16*)alloc((size_t)640 * 128 * 2);
    _Float16* redWh_p = (_Float16*)alloc((size_t)512 * 256 * 2);
    _Float16* redWc_p = (_Float16*)alloc((size_t)512 * 256 * 2);
    _Float16* out1W_p = (_Float16*)alloc((size_t)768 * 256 * 2);
    _Float16* out2W_p = (_Float16*)alloc((size_t)256 * V_ * 2);
    float* enc_out    = (float*)alloc((size_t)B_ * TENC_ * 512 * 4);
    _Float16* ench    = (_Float16*)alloc((size_t)B_ * TENC_ * 512 * 2);
    float* enc_feat   = (float*)alloc((size_t)B_ * TENC_ * 512 * 4);
    float* hfin = (float*)alloc(B_ * 512 * 4);
    float* cfin = (float*)alloc(B_ * 512 * 4);
    float* h    = (float*)alloc(B_ * 256 * 4);
    float* c    = (float*)alloc(B_ * 256 * 4);
    float* ctx  = (float*)alloc(B_ * 512 * 4);
    float* cov  = (float*)alloc(B_ * 512 * 4);
    float* attn = (float*)alloc(B_ * 512 * 4);
    float* x_in = (float*)alloc(B_ * 128 * 4);
    float* out1h = (float*)alloc(B_ * 256 * 4);
    _Float16* out1h_h = (_Float16*)alloc(B_ * 256 * 2);
    float* pgen     = (float*)alloc(B_ * 4);
    float* sumexp   = (float*)alloc(B_ * 4);
    float* loss_acc = (float*)alloc(B_ * 4);
    (void)ws_size; (void)in_sizes; (void)n_in; (void)out_size;

    auto blks = [](size_t n) { return (unsigned)((n + 255) / 256); };

    // --- Weight prep: pack into WMMA-fragment-major f16 --------------------
    pack_kernel<<<blks(640 * 128), 256, 0, stream>>>(xcW_p, xc_W, 640, 128, 0);
    pack_kernel<<<blks(512 * 512), 256, 0, stream>>>(attWh_p, att_Wh, 512, 512, 0);
    pack_kernel<<<blks(512 * 512), 256, 0, stream>>>(attWs_p, att_Ws, 512, 512, 0);
    pack_kernel<<<blks(512 * 256), 256, 0, stream>>>(redWh_p, red_Wh, 512, 256, 0);
    pack_kernel<<<blks(512 * 256), 256, 0, stream>>>(redWc_p, red_Wc, 512, 256, 0);
    pack_kernel<<<blks(768 * 256), 256, 0, stream>>>(out1W_p, out1_W, 768, 256, 0);
    pack_kernel<<<blks((size_t)256 * V_), 256, 0, stream>>>(out2W_p, out2_W, 256, V_, 0);
    // Gate weights: k-rows 0..127 = Wih^T (4 k-chunks = 128*1024 packed elems),
    // k-rows 128..383 = Whh^T. src is [N=1024][K] (PyTorch), so transposed=1.
    pack_kernel<<<blks(1024 * 128), 256, 0, stream>>>(Wenc,              enc_Wih_f, 128, 1024, 1);
    pack_kernel<<<blks(1024 * 256), 256, 0, stream>>>(Wenc + 128 * 1024, enc_Whh_f, 256, 1024, 1);
    pack_kernel<<<blks(1024 * 128), 256, 0, stream>>>(Wenc + 384 * 1024, enc_Wih_b, 128, 1024, 1);
    pack_kernel<<<blks(1024 * 256), 256, 0, stream>>>(Wenc + (384 + 128) * 1024, enc_Whh_b, 256, 1024, 1);
    pack_kernel<<<blks(1024 * 128), 256, 0, stream>>>(Wdec,              dec_Wih, 128, 1024, 1);
    pack_kernel<<<blks(1024 * 256), 256, 0, stream>>>(Wdec + 128 * 1024, dec_Whh, 256, 1024, 1);
    embA_kernel<<<blks((size_t)B_ * TENC_ * E_), 256, 0, stream>>>(embA, embed, article_ids, B_ * TENC_);

    // --- Encoder -----------------------------------------------------------
    enc_lstm_kernel<<<2, 1024, 0, stream>>>(embA, Wenc, enc_b_f, enc_b_b, enc_out, hfin, cfin);
    cvt_f16_kernel<<<blks((size_t)B_ * TENC_ * 512), 256, 0, stream>>>(ench, enc_out, B_ * TENC_ * 512);
    reduce_state_kernel<<<1, 1024, 0, stream>>>(hfin, cfin, redWh_p, red_bh, redWc_p, red_bc,
                                                h, c, h_context, coverage, ctx, cov, loss_acc);
    feat_kernel<<<dim3((B_ * TENC_) / 16, 4), 256, 0, stream>>>(ench, attWh_p, enc_feat);

    // --- Decoder loop ------------------------------------------------------
    for (int t = 0; t < TDEC_; ++t) {
        dec_step_kernel<<<1, 1024, 0, stream>>>(
            t, embed, sids, smask, amask, xc_b, xcW_p, dec_b, Wdec, att_b, attWs_p,
            att_wc, att_v, enc_feat, enc_out, pgen_W, pgen_b, out1_b, out1W_p,
            h, c, ctx, cov, attn, x_in, out1h, out1h_h, pgen, sumexp, loss_acc);
        vocab_kernel<<<(2 * (V_ / 16) + 7) / 8, 256, 0, stream>>>(out1h_h, out2W_p, out2_b, sumexp);
        step_loss_kernel<<<1, 1024, 0, stream>>>(t, tgt, smask, out1h, out2_W, out2_b,
                                                 pgen, sumexp, ext_ids, attn, loss_acc);
    }
    finalize_kernel<<<1, 32, 0, stream>>>(loss_acc, slen, (float*)d_out);
}
